// ROEN_Final_33526514713351
// MI455X (gfx1250) — matrix operand order repaired and checked by reading, not compiled, not run on hardware
//
#include <hip/hip_runtime.h>
#include <math.h>

#define HDIM   128
#define NHEADS 8
#define HD     16

typedef __attribute__((ext_vector_type(16))) __bf16 v16bf;
typedef __attribute__((ext_vector_type(8)))  float  v8f;

// ---------- helpers ----------

// monotonic float->uint encoding so float-max == uint-max
__device__ __forceinline__ unsigned fenc(float f) {
  unsigned u = __float_as_uint(f);
  return (u & 0x80000000u) ? ~u : (u | 0x80000000u);
}
__device__ __forceinline__ float fdec(unsigned u) {
  return (u & 0x80000000u) ? __uint_as_float(u & 0x7fffffffu)
                           : __uint_as_float(~u);
}
// bf16 pair unpack from a dword (bf16 -> f32 is a shift)
__device__ __forceinline__ float bflo(unsigned w) { return __uint_as_float(w << 16); }
__device__ __forceinline__ float bfhi(unsigned w) { return __uint_as_float(w & 0xffff0000u); }

// Load one 16x32 bf16 A-fragment from an fp32 row (ISA 7.12.2 16-bit A layout):
// lanes 0-15 hold row, elems 0-7 = K(kb*32+0..7), 8-15 = K(kb*32+16..23);
// lanes 16-31: +8 K offset.
__device__ __forceinline__ v16bf load_a_frag(const float* __restrict__ row, int kb, int lane) {
  const int off = kb * 32 + ((lane & 16) ? 8 : 0);
  const float4 x0 = *(const float4*)(row + off);
  const float4 x1 = *(const float4*)(row + off + 4);
  const float4 y0 = *(const float4*)(row + off + 16);
  const float4 y1 = *(const float4*)(row + off + 20);
  v16bf a;
  a[0] = (__bf16)x0.x; a[1] = (__bf16)x0.y; a[2]  = (__bf16)x0.z; a[3]  = (__bf16)x0.w;
  a[4] = (__bf16)x1.x; a[5] = (__bf16)x1.y; a[6]  = (__bf16)x1.z; a[7]  = (__bf16)x1.w;
  a[8] = (__bf16)y0.x; a[9] = (__bf16)y0.y; a[10] = (__bf16)y0.z; a[11] = (__bf16)y0.w;
  a[12]= (__bf16)y1.x; a[13]= (__bf16)y1.y; a[14] = (__bf16)y1.z; a[15] = (__bf16)y1.w;
  return a;
}

#define WMMA_BF16(A, B, C) \
  __builtin_amdgcn_wmma_f32_16x16x32_bf16(false, (A), false, (B), (short)0, (C), false, false)

// ---------- kernels ----------

// Pack a 128x128 fp32 weight (K-major rows) into per-lane-contiguous bf16
// B-fragments: P[((nt*4+kb)*32+lane)*16+i] = W[(kb*32+(lane>>4)*8+i+(i>=8?8:0))*128 + nt*16+(lane&15)]
__global__ void roen_pack_w(const float* __restrict__ W, __bf16* __restrict__ P) {
  int t = blockIdx.x * blockDim.x + threadIdx.x;
  if (t >= HDIM * HDIM) return;
  int i = t & 15, lane = (t >> 4) & 31, kb = (t >> 9) & 3, nt = t >> 11;
  int krow = kb * 32 + ((lane >> 4) & 1) * 8 + i + ((i >= 8) ? 8 : 0);
  int col  = nt * 16 + (lane & 15);
  P[t] = (__bf16)W[krow * HDIM + col];
}

__global__ void roen_init(unsigned* __restrict__ segmax, float* __restrict__ denom,
                          float* __restrict__ agg, float* __restrict__ csum,
                          float* __restrict__ csq, int n) {
  int t = blockIdx.x * blockDim.x + threadIdx.x;
  int nseg = n * NHEADS;
  if (t < nseg) { segmax[t] = 0x007fffffu; /* enc(-inf) */ denom[t] = 0.0f; }
  if (t < n * HDIM) agg[t] = 0.0f;
  if (t < HDIM) { csum[t] = 0.0f; csq[t] = 0.0f; }
}

// Q/K/V GEMM: one wave per 16-row tile, A fragments reused across 3 weights x 8 col tiles.
__global__ void roen_qkv_gemm(const float* __restrict__ x,
                              const __bf16* __restrict__ PQ, const __bf16* __restrict__ PK,
                              const __bf16* __restrict__ PV,
                              float* __restrict__ q, float* __restrict__ k,
                              float* __restrict__ v, int nrows) {
  const int wave = (blockIdx.x * blockDim.x + threadIdx.x) >> 5;
  const int lane = threadIdx.x & 31;
  const int m0 = wave * 16;
  if (m0 + 16 > nrows) return;
  const float* row = x + (size_t)(m0 + (lane & 15)) * HDIM;
  v16bf a0 = load_a_frag(row, 0, lane);
  v16bf a1 = load_a_frag(row, 1, lane);
  v16bf a2 = load_a_frag(row, 2, lane);
  v16bf a3 = load_a_frag(row, 3, lane);
  const __bf16* Ws[3] = {PQ, PK, PV};
  float* Os[3] = {q, k, v};
  const int col = lane & 15;
  const int rb  = m0 + ((lane & 16) ? 8 : 0);
  #pragma unroll
  for (int w = 0; w < 3; ++w) {
    const __bf16* P = Ws[w];
    float* O = Os[w];
    #pragma unroll
    for (int nt = 0; nt < 8; ++nt) {
      const __bf16* pb = P + nt * 2048 + lane * 16;
      v16bf b0 = *(const v16bf*)(pb);
      v16bf b1 = *(const v16bf*)(pb + 512);
      v16bf b2 = *(const v16bf*)(pb + 1024);
      v16bf b3 = *(const v16bf*)(pb + 1536);
      v8f c = {};
      c = WMMA_BF16(a0, b0, c);
      c = WMMA_BF16(a1, b1, c);
      c = WMMA_BF16(a2, b2, c);
      c = WMMA_BF16(a3, b3, c);
      #pragma unroll
      for (int i = 0; i < 8; ++i)
        O[(size_t)(rb + i) * HDIM + nt * 16 + col] = c[i];
    }
  }
}

// Edge-feature GEMM: e = edge_attr @ WE, output cached in bf16 (halves HBM re-read cost).
__global__ void roen_edge_gemm(const float* __restrict__ ea, const __bf16* __restrict__ PE,
                               __bf16* __restrict__ e, int nrows) {
  const int wave = (blockIdx.x * blockDim.x + threadIdx.x) >> 5;
  const int lane = threadIdx.x & 31;
  const int m0 = wave * 16;
  if (m0 + 16 > nrows) return;
  const float* row = ea + (size_t)(m0 + (lane & 15)) * HDIM;
  v16bf a0 = load_a_frag(row, 0, lane);
  v16bf a1 = load_a_frag(row, 1, lane);
  v16bf a2 = load_a_frag(row, 2, lane);
  v16bf a3 = load_a_frag(row, 3, lane);
  const int col = lane & 15;
  const int rb  = m0 + ((lane & 16) ? 8 : 0);
  #pragma unroll
  for (int nt = 0; nt < 8; ++nt) {
    const __bf16* pb = PE + nt * 2048 + lane * 16;
    v16bf b0 = *(const v16bf*)(pb);
    v16bf b1 = *(const v16bf*)(pb + 512);
    v16bf b2 = *(const v16bf*)(pb + 1024);
    v16bf b3 = *(const v16bf*)(pb + 1536);
    v8f c = {};
    c = WMMA_BF16(a0, b0, c);
    c = WMMA_BF16(a1, b1, c);
    c = WMMA_BF16(a2, b2, c);
    c = WMMA_BF16(a3, b3, c);
    #pragma unroll
    for (int i = 0; i < 8; ++i)
      e[(size_t)(rb + i) * HDIM + nt * 16 + col] = (__bf16)c[i];
  }
}

// score[edge,h] = 0.25 * sum_i q[dst][h*16+i] * (k[src][h*16+i] + e[edge][h*16+i])
// + running segment max via encoded atomic u32 max.
__global__ void roen_score(const float* __restrict__ q, const float* __restrict__ k,
                           const __bf16* __restrict__ e, const int* __restrict__ ei,
                           int E, float* __restrict__ score, unsigned* __restrict__ segmax) {
  int t = blockIdx.x * blockDim.x + threadIdx.x;
  if (t >= E * NHEADS) return;
  int edge = t >> 3, h = t & 7;
  int s = ei[edge], d = ei[E + edge];
  const float4* qr = (const float4*)(q + (size_t)d * HDIM + h * HD);
  const float4* kr = (const float4*)(k + (size_t)s * HDIM + h * HD);
  const uint4*  er = (const uint4*)(e + (size_t)edge * HDIM + h * HD);
  float acc = 0.0f;
  #pragma unroll
  for (int j = 0; j < 2; ++j) {
    uint4 ew = er[j];
    float4 q0 = qr[2 * j], q1 = qr[2 * j + 1];
    float4 k0 = kr[2 * j], k1 = kr[2 * j + 1];
    acc += q0.x * (k0.x + bflo(ew.x)) + q0.y * (k0.y + bfhi(ew.x));
    acc += q0.z * (k0.z + bflo(ew.y)) + q0.w * (k0.w + bfhi(ew.y));
    acc += q1.x * (k1.x + bflo(ew.z)) + q1.y * (k1.y + bfhi(ew.z));
    acc += q1.z * (k1.z + bflo(ew.w)) + q1.w * (k1.w + bfhi(ew.w));
  }
  acc *= 0.25f;  // 1/sqrt(HD)
  score[t] = acc;
  atomicMax(segmax + d * NHEADS + h, fenc(acc));
}

__global__ void roen_expdenom(float* __restrict__ score, const int* __restrict__ ei,
                              const unsigned* __restrict__ segmax,
                              float* __restrict__ denom, int E) {
  int t = blockIdx.x * blockDim.x + threadIdx.x;
  if (t >= E * NHEADS) return;
  int edge = t >> 3, h = t & 7;
  int d = ei[E + edge];
  float m = fdec(segmax[d * NHEADS + h]);
  if (!(m >= -3.0e38f && m <= 3.0e38f)) m = 0.0f;  // reference: non-finite -> 0
  float ex = __expf(score[t] - m);
  score[t] = ex;
  atomicAdd(denom + d * NHEADS + h, ex);
}

__global__ void roen_message(const float* __restrict__ ex, const float* __restrict__ denom,
                             const float* __restrict__ v, const __bf16* __restrict__ e,
                             const int* __restrict__ ei, float* __restrict__ agg, int E) {
  int t = blockIdx.x * blockDim.x + threadIdx.x;
  if (t >= E * NHEADS) return;
  int edge = t >> 3, h = t & 7;
  int s = ei[edge], d = ei[E + edge];
  float alpha = ex[t] / (denom[d * NHEADS + h] + 1e-16f);
  const float4* vr = (const float4*)(v + (size_t)s * HDIM + h * HD);
  const uint4*  er = (const uint4*)(e + (size_t)edge * HDIM + h * HD);
  float* ab = agg + (size_t)d * HDIM + h * HD;
  float vals[16];
  #pragma unroll
  for (int j = 0; j < 2; ++j) {
    uint4 ew = er[j];
    float4 v0 = vr[2 * j], v1 = vr[2 * j + 1];
    vals[8 * j + 0] = alpha * (v0.x + bflo(ew.x));
    vals[8 * j + 1] = alpha * (v0.y + bfhi(ew.x));
    vals[8 * j + 2] = alpha * (v0.z + bflo(ew.y));
    vals[8 * j + 3] = alpha * (v0.w + bfhi(ew.y));
    vals[8 * j + 4] = alpha * (v1.x + bflo(ew.z));
    vals[8 * j + 5] = alpha * (v1.y + bfhi(ew.z));
    vals[8 * j + 6] = alpha * (v1.z + bflo(ew.w));
    vals[8 * j + 7] = alpha * (v1.w + bfhi(ew.w));
  }
  #pragma unroll
  for (int i = 0; i < 16; ++i) atomicAdd(ab + i, vals[i]);
}

// h = agg@Wout + bout + x; also accumulate per-column sum(h), sum(h^2) for GraphNorm.
__global__ void roen_out_gemm(const float* __restrict__ agg, const __bf16* __restrict__ PO,
                              const float* __restrict__ bout, const float* __restrict__ x,
                              float* __restrict__ h, float* __restrict__ csum,
                              float* __restrict__ csq, int nrows) {
  const int wave = (blockIdx.x * blockDim.x + threadIdx.x) >> 5;
  const int lane = threadIdx.x & 31;
  const int m0 = wave * 16;
  if (m0 + 16 > nrows) return;
  const float* row = agg + (size_t)(m0 + (lane & 15)) * HDIM;
  v16bf a0 = load_a_frag(row, 0, lane);
  v16bf a1 = load_a_frag(row, 1, lane);
  v16bf a2 = load_a_frag(row, 2, lane);
  v16bf a3 = load_a_frag(row, 3, lane);
  const int rb = m0 + ((lane & 16) ? 8 : 0);
  #pragma unroll
  for (int nt = 0; nt < 8; ++nt) {
    const __bf16* pb = PO + nt * 2048 + lane * 16;
    v16bf b0 = *(const v16bf*)(pb);
    v16bf b1 = *(const v16bf*)(pb + 512);
    v16bf b2 = *(const v16bf*)(pb + 1024);
    v16bf b3 = *(const v16bf*)(pb + 1536);
    v8f c = {};
    c = WMMA_BF16(a0, b0, c);
    c = WMMA_BF16(a1, b1, c);
    c = WMMA_BF16(a2, b2, c);
    c = WMMA_BF16(a3, b3, c);
    const int col = nt * 16 + (lane & 15);
    const float bo = bout[col];
    float s1 = 0.0f, s2 = 0.0f;
    #pragma unroll
    for (int i = 0; i < 8; ++i) {
      size_t idx = (size_t)(rb + i) * HDIM + col;
      float hv = c[i] + bo + x[idx];
      h[idx] = hv;
      s1 += hv;
      s2 += hv * hv;
    }
    atomicAdd(csum + col, s1);
    atomicAdd(csq + col, s2);
  }
}

// var(h - m*ms) = E[h^2] - 2*ms*m^2 + ms^2*m^2  (m = E[h]) -> single-pass stats.
__global__ void roen_stats(const float* __restrict__ csum, const float* __restrict__ csq,
                           const float* __restrict__ gn_w, const float* __restrict__ gn_ms,
                           float* __restrict__ scale, float* __restrict__ sub, int n) {
  int c = threadIdx.x;
  if (c >= HDIM) return;
  float inv = 1.0f / (float)n;
  float m  = csum[c] * inv;
  float q2 = csq[c] * inv;
  float ms = gn_ms[c];
  float var = q2 - 2.0f * ms * m * m + ms * ms * m * m;
  scale[c] = gn_w[c] * rsqrtf(var + 1e-5f);
  sub[c]   = m * ms;
}

__global__ void roen_norm_gelu(const float* __restrict__ h, const float* __restrict__ scale,
                               const float* __restrict__ sub, const float* __restrict__ gn_b,
                               float* __restrict__ out, int total) {
  int t = blockIdx.x * blockDim.x + threadIdx.x;
  if (t >= total) return;
  int c = t & (HDIM - 1);
  float o = scale[c] * (h[t] - sub[c]) + gn_b[c];
  out[t] = 0.5f * o * (1.0f + erff(o * 0.70710678118654752f));  // exact GELU
}

// ---------- launch ----------

extern "C" void kernel_launch(void* const* d_in, const int* in_sizes, int n_in,
                              void* d_out, int out_size, void* d_ws, size_t ws_size,
                              hipStream_t stream) {
  (void)n_in; (void)out_size; (void)ws_size;
  const float* x    = (const float*)d_in[0];
  const int*   ei   = (const int*)d_in[1];
  const float* ea   = (const float*)d_in[2];
  const float* WQ   = (const float*)d_in[3];
  const float* WK   = (const float*)d_in[4];
  const float* WV   = (const float*)d_in[5];
  const float* WE   = (const float*)d_in[6];
  const float* WO   = (const float*)d_in[7];
  const float* bo   = (const float*)d_in[8];
  const float* gnw  = (const float*)d_in[9];
  const float* gnb  = (const float*)d_in[10];
  const float* gnms = (const float*)d_in[11];
  float* out = (float*)d_out;

  const int n = in_sizes[0] / HDIM;   // 20000
  const int E = in_sizes[2] / HDIM;   // 640000

  char* w = (char*)d_ws;
  auto take = [&](size_t bytes) -> char* {
    char* p = w;
    w += (bytes + 255) & ~(size_t)255;
    return p;
  };
  float*    qb   = (float*)take((size_t)n * HDIM * 4);
  float*    kb   = (float*)take((size_t)n * HDIM * 4);
  float*    vb   = (float*)take((size_t)n * HDIM * 4);
  __bf16*   ebf  = (__bf16*)take((size_t)E * HDIM * 2);
  float*    sc   = (float*)take((size_t)E * NHEADS * 4);
  unsigned* smax = (unsigned*)take((size_t)n * NHEADS * 4);
  float*    den  = (float*)take((size_t)n * NHEADS * 4);
  float*    agg  = (float*)take((size_t)n * HDIM * 4);
  float*    hbuf = (float*)take((size_t)n * HDIM * 4);
  __bf16*   PQ   = (__bf16*)take(HDIM * HDIM * 2);
  __bf16*   PK   = (__bf16*)take(HDIM * HDIM * 2);
  __bf16*   PV   = (__bf16*)take(HDIM * HDIM * 2);
  __bf16*   PE   = (__bf16*)take(HDIM * HDIM * 2);
  __bf16*   PO   = (__bf16*)take(HDIM * HDIM * 2);
  float*    csum = (float*)take(HDIM * 4);
  float*    csq  = (float*)take(HDIM * 4);
  float*    scl  = (float*)take(HDIM * 4);
  float*    sub  = (float*)take(HDIM * 4);

  const int TB = 256;
  const int pg = (HDIM * HDIM + TB - 1) / TB;
  roen_pack_w<<<pg, TB, 0, stream>>>(WQ, PQ);
  roen_pack_w<<<pg, TB, 0, stream>>>(WK, PK);
  roen_pack_w<<<pg, TB, 0, stream>>>(WV, PV);
  roen_pack_w<<<pg, TB, 0, stream>>>(WE, PE);
  roen_pack_w<<<pg, TB, 0, stream>>>(WO, PO);

  roen_init<<<((size_t)n * HDIM + TB - 1) / TB, TB, 0, stream>>>(smax, den, agg, csum, csq, n);

  const int nwaves = n / 16;
  const int nblk = (nwaves * 32 + TB - 1) / TB;
  roen_qkv_gemm<<<nblk, TB, 0, stream>>>(x, PQ, PK, PV, qb, kb, vb, n);

  const int ewaves = E / 16;
  roen_edge_gemm<<<(ewaves * 32 + TB - 1) / TB, TB, 0, stream>>>(ea, PE, ebf, E);

  const int et = E * NHEADS;
  const int eblk = (et + TB - 1) / TB;
  roen_score<<<eblk, TB, 0, stream>>>(qb, kb, ebf, ei, E, sc, smax);
  roen_expdenom<<<eblk, TB, 0, stream>>>(sc, ei, smax, den, E);
  roen_message<<<eblk, TB, 0, stream>>>(sc, den, vb, ebf, ei, agg, E);

  roen_out_gemm<<<nblk, TB, 0, stream>>>(agg, PO, bo, x, hbuf, csum, csq, n);
  roen_stats<<<1, 128, 0, stream>>>(csum, csq, gnw, gnms, scl, sub, n);
  roen_norm_gelu<<<((size_t)n * HDIM + TB - 1) / TB, TB, 0, stream>>>(hbuf, scl, sub, gnb, out, n * HDIM);
}
